// GraphNetwork_58153857188014
// MI455X (gfx1250) — matrix operand order repaired and checked by reading, not compiled, not run on hardware
//
#include <hip/hip_runtime.h>

// ---------------------------------------------------------------------------
// Types for CDNA5 WMMA (wave32, v_wmma_f32_16x16x32_bf16)
// ---------------------------------------------------------------------------
typedef __bf16 bf16_t;
typedef __attribute__((ext_vector_type(16))) __bf16    v16bf;
typedef __attribute__((ext_vector_type(8)))  float     v8f;
typedef __attribute__((ext_vector_type(4)))  unsigned  u32x4;

union ABv { u32x4 q[2]; v16bf v; };   // 32 bytes per lane = one 16x32 bf16 operand

// LDS activation buffers (per-wave private regions), carved from dynamic LDS
constexpr int S1   = 264;  // bf16 elems/row, buffer1 (K<=256; 528B row, 16B mult)
constexpr int S2   = 136;  // bf16 elems/row, buffer2 (K=128; 272B row, 16B mult)
constexpr int FS1  = 132;  // f32 elems/row when buffer1 is viewed as float
constexpr int WAVES = 4;   // 128 threads/block = 4 wave32

constexpr int SZ_A  = WAVES * 16 * S1 * 2;   // 33792 B
constexpr int SZ_B  = WAVES * 16 * S2 * 2;   // 17408 B
constexpr int SZ_ST = WAVES * 32 * 4;        //   512 B
constexpr int WOFF  = SZ_A + SZ_B + SZ_ST;   // 51712 B (16-aligned)
constexpr int SMEM_ENC  = WOFF + 32 * 1024;  // biggest layer = 32KB packed
constexpr int SMEM_PROC = WOFF + 64 * 1024;  // proc layer0 = 64KB packed

#define SMEM_CARVE                                              \
  extern __shared__ char smem[];                                \
  bf16_t* sA      = (bf16_t*)(smem);                            \
  bf16_t* sBb     = (bf16_t*)(smem + SZ_A);                     \
  float*  sStats  = (float*)(smem + SZ_A + SZ_B);               \
  bf16_t* sW      = (bf16_t*)(smem + WOFF);

// ---------------------------------------------------------------------------
// Async stage of packed weights (global, contiguous) into LDS.
// Emits GLOBAL_LOAD_ASYNC_TO_LDS_B128 (ASYNCcnt) per 16B chunk per thread,
// then s_wait_asynccnt 0 + block barrier. bytes must be a multiple of 2048.
// ---------------------------------------------------------------------------
__device__ __forceinline__ void stage_weights_async(const unsigned char* gsrc,
                                                    bf16_t* ldst, int bytes) {
  unsigned lds0 = (unsigned)(uintptr_t)ldst;   // low 32 bits = LDS byte address
  for (int off = (int)threadIdx.x * 16; off < bytes; off += 128 * 16) {
    unsigned ldsa = lds0 + (unsigned)off;
    asm volatile("global_load_async_to_lds_b128 %0, %1, %2"
                 :: "v"(ldsa), "v"(off), "s"(gsrc)
                 : "memory");
  }
  asm volatile("s_wait_asynccnt 0" ::: "memory");
  __syncthreads();
}

// ---------------------------------------------------------------------------
// One MLP layer on a 16-row block: D = relu(A[16xK] * W[Kx(16*NTS)] + b)
// A in LDS (ISA 16-bit A layout; 2x ds_load_b128), W staged in LDS in the ISA
// 16-bit B layout (2x ds_load_b128 per lane per tile), bias folded into C.
// ---------------------------------------------------------------------------
template <int KTS, int NTS, bool RELU>
__device__ __forceinline__ void mlp_layer_bf16(const bf16_t* in, int inS,
                                               bf16_t* out, int outS,
                                               const bf16_t* sW,
                                               const float* __restrict__ bias) {
  const int lane = threadIdx.x & 31;
  const int m = lane & 15;        // A row / D col within tile
  const int g = lane >> 4;        // K-half selector (ISA A layout)
  ABv a[KTS];
#pragma unroll
  for (int kt = 0; kt < KTS; ++kt) {
    const bf16_t* ap = in + m * inS + kt * 32 + g * 8;
    a[kt].q[0] = *(const u32x4*)(ap);        // K = kt*32 + 8g   .. +7
    a[kt].q[1] = *(const u32x4*)(ap + 16);   // K = kt*32 +16+8g .. +7
  }
#pragma unroll
  for (int nt = 0; nt < NTS; ++nt) {
    float bc = bias[nt * 16 + m];
    v8f acc = {bc, bc, bc, bc, bc, bc, bc, bc};
#pragma unroll
    for (int kt = 0; kt < KTS; ++kt) {
      ABv b;
      const u32x4* bp = (const u32x4*)sW + (size_t)((nt * KTS + kt) * 32 + lane) * 2;
      b.q[0] = bp[0];
      b.q[1] = bp[1];
      acc = __builtin_amdgcn_wmma_f32_16x16x32_bf16(
          false, a[kt].v, false, b.v, (short)0, acc, false, false);
    }
#pragma unroll
    for (int v = 0; v < 8; ++v) {
      float x = RELU ? fmaxf(acc[v], 0.f) : acc[v];
      out[(v + 8 * g) * outS + nt * 16 + m] = (bf16_t)x;   // D: row=v+8g, col=m
    }
  }
}

// Final layer variant: f32 output into LDS (for LayerNorm / decoder readout)
template <int KTS, int NTS>
__device__ __forceinline__ void mlp_layer_f32out(const bf16_t* in, int inS,
                                                 float* out, int outS,
                                                 const bf16_t* sW,
                                                 const float* __restrict__ bias) {
  const int lane = threadIdx.x & 31;
  const int m = lane & 15;
  const int g = lane >> 4;
  ABv a[KTS];
#pragma unroll
  for (int kt = 0; kt < KTS; ++kt) {
    const bf16_t* ap = in + m * inS + kt * 32 + g * 8;
    a[kt].q[0] = *(const u32x4*)(ap);
    a[kt].q[1] = *(const u32x4*)(ap + 16);
  }
#pragma unroll
  for (int nt = 0; nt < NTS; ++nt) {
    float bc = bias[nt * 16 + m];
    v8f acc = {bc, bc, bc, bc, bc, bc, bc, bc};
#pragma unroll
    for (int kt = 0; kt < KTS; ++kt) {
      ABv b;
      const u32x4* bp = (const u32x4*)sW + (size_t)((nt * KTS + kt) * 32 + lane) * 2;
      b.q[0] = bp[0];
      b.q[1] = bp[1];
      acc = __builtin_amdgcn_wmma_f32_16x16x32_bf16(
          false, a[kt].v, false, b.v, (short)0, acc, false, false);
    }
#pragma unroll
    for (int v = 0; v < 8; ++v)
      out[(v + 8 * g) * outS + nt * 16 + m] = acc[v];
  }
}

// 4-layer MLP: A1(K0) -> A2 -> A1 -> A2 -> f32 into A1.
// Weights staged layer-by-layer through the shared LDS weight buffer.
template <int K0, int LASTNT>
__device__ __forceinline__ void mlp4_wave(bf16_t* A1, bf16_t* A2, bf16_t* sW,
                                          const unsigned char* __restrict__ wg,
                                          const float* b0, const float* b1,
                                          const float* b2, const float* b3) {
  constexpr int L0B = (K0 / 32) * 8 * 1024;   // layer-0 packed bytes
  constexpr int L1B = 4 * 8 * 1024;           // 128-K layer packed bytes
  __builtin_prefetch(wg + L0B, 0, 0);         // global_prefetch next layer
  stage_weights_async(wg, sW, L0B);
  mlp_layer_bf16<K0 / 32, 8, true>(A1, S1, A2, S2, sW, b0);
  __syncthreads();                             // all waves done reading sW
  stage_weights_async(wg + L0B, sW, L1B);
  mlp_layer_bf16<4, 8, true>(A2, S2, A1, S1, sW, b1);
  __syncthreads();
  stage_weights_async(wg + L0B + L1B, sW, L1B);
  mlp_layer_bf16<4, 8, true>(A1, S1, A2, S2, sW, b2);
  __syncthreads();
  stage_weights_async(wg + L0B + 2 * L1B, sW, L1B);
  mlp_layer_f32out<4, LASTNT>(A2, S2, (float*)A1, FS1, sW, b3);
  __syncthreads();
}

// Row-wise LayerNorm stats over 128 cols; both lane halves reduce 64 cols
// each, combined with a wave32 xor-shuffle.
__device__ __forceinline__ void ln_stats_wave(const float* f, float* stats) {
  const int lane = threadIdx.x & 31;
  const int r = lane & 15;
  const float* p = f + r * FS1 + (lane >> 4) * 64;
  float s = 0.f, s2 = 0.f;
#pragma unroll 4
  for (int c = 0; c < 64; ++c) {
    float x = p[c];
    s += x;
    s2 = fmaf(x, x, s2);
  }
  s += __shfl_xor(s, 16);
  s2 += __shfl_xor(s2, 16);
  if (lane < 16) {
    float mu = s * 0.0078125f;
    float var = s2 * 0.0078125f - mu * mu;
    stats[2 * r] = mu;
    stats[2 * r + 1] = rsqrtf(var + 1e-5f);
  }
  __syncthreads();
}

// ---------------------------------------------------------------------------
// Weight repacking: fp32 row-major [K,N] -> bf16 tiles in ISA 16-bit B layout.
// tile t = nt*kts + kt; dword d of lane holds (k = kt*32 + (d>>2)*16 + (lane>>4)*8 + 2(d&3)(+1),
// n = nt*16 + (lane&15)). Zero-padded to Kpad/Npad.
// ---------------------------------------------------------------------------
__global__ void k_pack(const float* __restrict__ W, int Kreal, int Nreal,
                       int kts, int nts, unsigned* __restrict__ out) {
  int id = blockIdx.x * 256 + threadIdx.x;
  int total = kts * nts * 256;
  if (id >= total) return;
  int t = id >> 8, w = id & 255;
  int lane = w >> 3, d = w & 7;
  int kt = t % kts, nt = t / kts;
  int k0 = kt * 32 + (d >> 2) * 16 + (lane >> 4) * 8 + 2 * (d & 3);
  int n = nt * 16 + (lane & 15);
  float f0 = (k0 < Kreal && n < Nreal) ? W[(size_t)k0 * Nreal + n] : 0.f;
  float f1 = (k0 + 1 < Kreal && n < Nreal) ? W[(size_t)(k0 + 1) * Nreal + n] : 0.f;
  unsigned short u0 = __builtin_bit_cast(unsigned short, (bf16_t)f0);
  unsigned short u1 = __builtin_bit_cast(unsigned short, (bf16_t)f1);
  out[id] = (unsigned)u0 | ((unsigned)u1 << 16);
}

// ---------------------------------------------------------------------------
// Encoder (node): concat(x[25], embed[ptype][16]) pad->64, MLP+LN -> nf
// ---------------------------------------------------------------------------
__global__ __launch_bounds__(128) void k_encode_node(
    const float* __restrict__ x, const int* __restrict__ ptype,
    const float* __restrict__ emb, const unsigned char* __restrict__ wg,
    const float* b0, const float* b1, const float* b2, const float* b3,
    const float* gw, const float* bw, float* __restrict__ nf, int N) {
  SMEM_CARVE
  const int w = threadIdx.x >> 5, lane = threadIdx.x & 31;
  const int row0 = (blockIdx.x * WAVES + w) * 16;
  bf16_t* A1 = sA + w * (16 * S1);
  bf16_t* A2 = sBb + w * (16 * S2);
  float* st = sStats + w * 32;
  for (int i = lane; i < 16 * 64; i += 32) {
    int r = i >> 6, c = i & 63;
    int n = row0 + r;
    if (n >= N) n = N - 1;
    float v = 0.f;
    if (c < 25)      v = x[(size_t)n * 25 + c];
    else if (c < 41) v = emb[ptype[n] * 16 + (c - 25)];
    A1[r * S1 + c] = (bf16_t)v;
  }
  __syncthreads();
  mlp4_wave<64, 8>(A1, A2, sW, wg, b0, b1, b2, b3);
  float* f = (float*)A1;
  ln_stats_wave(f, st);
  for (int i = lane; i < 2048; i += 32) {
    int r = i >> 7, c = i & 127;
    int n = row0 + r;
    if (n < N)
      nf[(size_t)n * 128 + c] =
          (f[r * FS1 + c] - st[2 * r]) * st[2 * r + 1] * gw[c] + bw[c];
  }
}

// Encoder (edge): edge_attr[4] pad->32, MLP+LN -> ef
__global__ __launch_bounds__(128) void k_encode_edge(
    const float* __restrict__ eattr, const unsigned char* __restrict__ wg,
    const float* b0, const float* b1, const float* b2, const float* b3,
    const float* gw, const float* bw, float* __restrict__ ef, int E) {
  SMEM_CARVE
  const int w = threadIdx.x >> 5, lane = threadIdx.x & 31;
  const int row0 = (blockIdx.x * WAVES + w) * 16;
  bf16_t* A1 = sA + w * (16 * S1);
  bf16_t* A2 = sBb + w * (16 * S2);
  float* st = sStats + w * 32;
  for (int i = lane; i < 16 * 32; i += 32) {
    int r = i >> 5, c = i & 31;
    int e = row0 + r;
    if (e >= E) e = E - 1;
    float v = (c < 4) ? eattr[(size_t)e * 4 + c] : 0.f;
    A1[r * S1 + c] = (bf16_t)v;
  }
  __syncthreads();
  mlp4_wave<32, 8>(A1, A2, sW, wg, b0, b1, b2, b3);
  float* f = (float*)A1;
  ln_stats_wave(f, st);
  for (int i = lane; i < 2048; i += 32) {
    int r = i >> 7, c = i & 127;
    int e = row0 + r;
    if (e < E)
      ef[(size_t)e * 128 + c] =
          (f[r * FS1 + c] - st[2 * r]) * st[2 * r + 1] * gw[c] + bw[c];
  }
}

// Processor edge step: new_ef = MLP_LN([ef, nf[src]]); agg += new_ef (atomic
// segment_sum on src); ef += new_ef
__global__ __launch_bounds__(128) void k_proc_edge(
    float* ef, const float* __restrict__ nf, float* agg,
    const int* __restrict__ src, const unsigned char* __restrict__ wg,
    const float* b0, const float* b1, const float* b2, const float* b3,
    const float* gw, const float* bw, int E) {
  SMEM_CARVE
  const int w = threadIdx.x >> 5, lane = threadIdx.x & 31;
  const int row0 = (blockIdx.x * WAVES + w) * 16;
  bf16_t* A1 = sA + w * (16 * S1);
  bf16_t* A2 = sBb + w * (16 * S2);
  float* st = sStats + w * 32;
  for (int i = lane; i < 16 * 256; i += 32) {
    int r = i >> 8, c = i & 255;
    int e = row0 + r;
    if (e >= E) e = E - 1;
    float v = (c < 128) ? ef[(size_t)e * 128 + c]
                        : nf[(size_t)src[e] * 128 + (c - 128)];
    A1[r * S1 + c] = (bf16_t)v;
  }
  __syncthreads();
  mlp4_wave<256, 8>(A1, A2, sW, wg, b0, b1, b2, b3);
  float* f = (float*)A1;
  ln_stats_wave(f, st);
  for (int i = lane; i < 2048; i += 32) {
    int r = i >> 7, c = i & 127;
    int e = row0 + r;
    if (e < E) {
      float val = (f[r * FS1 + c] - st[2 * r]) * st[2 * r + 1] * gw[c] + bw[c];
      atomicAdd(&agg[(size_t)src[e] * 128 + c], val);   // segment_sum
      ef[(size_t)e * 128 + c] += val;                   // residual
    }
  }
}

// Processor node step: new_nf = MLP_LN([nf, agg]); nf += new_nf
__global__ __launch_bounds__(128) void k_proc_node(
    float* nf, const float* __restrict__ agg,
    const unsigned char* __restrict__ wg,
    const float* b0, const float* b1, const float* b2, const float* b3,
    const float* gw, const float* bw, int N) {
  SMEM_CARVE
  const int w = threadIdx.x >> 5, lane = threadIdx.x & 31;
  const int row0 = (blockIdx.x * WAVES + w) * 16;
  bf16_t* A1 = sA + w * (16 * S1);
  bf16_t* A2 = sBb + w * (16 * S2);
  float* st = sStats + w * 32;
  for (int i = lane; i < 16 * 256; i += 32) {
    int r = i >> 8, c = i & 255;
    int n = row0 + r;
    if (n >= N) n = N - 1;
    float v = (c < 128) ? nf[(size_t)n * 128 + c]
                        : agg[(size_t)n * 128 + (c - 128)];
    A1[r * S1 + c] = (bf16_t)v;
  }
  __syncthreads();
  mlp4_wave<256, 8>(A1, A2, sW, wg, b0, b1, b2, b3);
  float* f = (float*)A1;
  ln_stats_wave(f, st);
  for (int i = lane; i < 2048; i += 32) {
    int r = i >> 7, c = i & 127;
    int n = row0 + r;
    if (n < N)
      nf[(size_t)n * 128 + c] +=
          (f[r * FS1 + c] - st[2 * r]) * st[2 * r + 1] * gw[c] + bw[c];
  }
}

// Decoder: 128->128->128->128->3 (no LN), last layer padded to one 16-col tile
__global__ __launch_bounds__(128) void k_decode(
    const float* __restrict__ nf, const unsigned char* __restrict__ wg,
    const float* b0, const float* b1, const float* b2, const float* b3,
    float* __restrict__ out, int N) {
  SMEM_CARVE
  const int w = threadIdx.x >> 5, lane = threadIdx.x & 31;
  const int row0 = (blockIdx.x * WAVES + w) * 16;
  bf16_t* A1 = sA + w * (16 * S1);
  bf16_t* A2 = sBb + w * (16 * S2);
  for (int i = lane; i < 16 * 128; i += 32) {
    int r = i >> 7, c = i & 127;
    int n = row0 + r;
    if (n >= N) n = N - 1;
    A1[r * S1 + c] = (bf16_t)nf[(size_t)n * 128 + c];
  }
  __syncthreads();
  mlp4_wave<128, 1>(A1, A2, sW, wg, b0, b1, b2, b3);
  float* f = (float*)A1;
  for (int i = lane; i < 48; i += 32) {
    int r = i / 3, c = i - r * 3;
    int n = row0 + r;
    if (n < N) out[(size_t)n * 3 + c] = f[r * FS1 + c];
  }
}

// ---------------------------------------------------------------------------
// Host orchestration.
// Input order (jax pytree: dict keys sorted): x, edge_index, edge_attr,
// particle_types, embed_table, enc{edge:W0..W3,b0..b3,be,g; node:...},
// proc{edge stacked; node stacked}, dec{W0..W3,b0..b3}.
// ---------------------------------------------------------------------------
enum {
  IN_X = 0, IN_EIDX = 1, IN_EATTR = 2, IN_PTYPE = 3, IN_EMB = 4,
  ENC_EDGE = 5,    // W0..W3(5..8) b0..b3(9..12) be(13) g(14)
  ENC_NODE = 15,   // 15..24
  PROC_EDGE = 25,  // stacked [10,...]: 25..34
  PROC_NODE = 35,  // 35..44
  DEC = 45         // W0..W3(45..48) b0..b3(49..52)
};

extern "C" void kernel_launch(void* const* d_in, const int* in_sizes, int n_in,
                              void* d_out, int out_size, void* d_ws, size_t ws_size,
                              hipStream_t stream) {
  const int N = in_sizes[IN_X] / 25;
  const int E = in_sizes[IN_EIDX] / 2;

  // Workspace layout
  float* nf = (float*)d_ws;
  float* agg = nf + (size_t)N * 128;
  float* ef = agg + (size_t)N * 128;
  unsigned* wbase = (unsigned*)(ef + (size_t)E * 128);

  // ---- pack all weights (bf16, B-layout) ----
  size_t cur = 0;
  auto pack1 = [&](const float* W, int Kreal, int Kpad, int Nreal, int Npad) {
    int kts = Kpad / 32, nts = Npad / 16;
    int total = kts * nts * 256;
    k_pack<<<(total + 255) / 256, 256, 0, stream>>>(W, Kreal, Nreal, kts, nts,
                                                    wbase + cur);
    cur += (size_t)total;
  };
  auto F = [&](int i) { return (const float*)d_in[i]; };

  size_t o_encE = cur;
  pack1(F(ENC_EDGE + 0), 4, 32, 128, 128);
  pack1(F(ENC_EDGE + 1), 128, 128, 128, 128);
  pack1(F(ENC_EDGE + 2), 128, 128, 128, 128);
  pack1(F(ENC_EDGE + 3), 128, 128, 128, 128);

  size_t o_encN = cur;
  pack1(F(ENC_NODE + 0), 41, 64, 128, 128);
  pack1(F(ENC_NODE + 1), 128, 128, 128, 128);
  pack1(F(ENC_NODE + 2), 128, 128, 128, 128);
  pack1(F(ENC_NODE + 3), 128, 128, 128, 128);

  size_t o_pE[10], o_pN[10];
  for (int s = 0; s < 10; ++s) {
    o_pE[s] = cur;
    pack1(F(PROC_EDGE + 0) + (size_t)s * 256 * 128, 256, 256, 128, 128);
    pack1(F(PROC_EDGE + 1) + (size_t)s * 128 * 128, 128, 128, 128, 128);
    pack1(F(PROC_EDGE + 2) + (size_t)s * 128 * 128, 128, 128, 128, 128);
    pack1(F(PROC_EDGE + 3) + (size_t)s * 128 * 128, 128, 128, 128, 128);
  }
  for (int s = 0; s < 10; ++s) {
    o_pN[s] = cur;
    pack1(F(PROC_NODE + 0) + (size_t)s * 256 * 128, 256, 256, 128, 128);
    pack1(F(PROC_NODE + 1) + (size_t)s * 128 * 128, 128, 128, 128, 128);
    pack1(F(PROC_NODE + 2) + (size_t)s * 128 * 128, 128, 128, 128, 128);
    pack1(F(PROC_NODE + 3) + (size_t)s * 128 * 128, 128, 128, 128, 128);
  }
  size_t o_dec = cur;
  pack1(F(DEC + 0), 128, 128, 128, 128);
  pack1(F(DEC + 1), 128, 128, 128, 128);
  pack1(F(DEC + 2), 128, 128, 128, 128);
  pack1(F(DEC + 3), 128, 128, 3, 16);

  const int blocksN = (N + 63) / 64;
  const int blocksE = (E + 63) / 64;
  auto WP = [&](size_t off) { return (const unsigned char*)(wbase + off); };

  // ---- encoders ----
  k_encode_edge<<<blocksE, 128, SMEM_ENC, stream>>>(
      F(IN_EATTR), WP(o_encE), F(ENC_EDGE + 4), F(ENC_EDGE + 5),
      F(ENC_EDGE + 6), F(ENC_EDGE + 7), F(ENC_EDGE + 9) /*g*/,
      F(ENC_EDGE + 8) /*be*/, ef, E);
  k_encode_node<<<blocksN, 128, SMEM_ENC, stream>>>(
      F(IN_X), (const int*)d_in[IN_PTYPE], F(IN_EMB), WP(o_encN),
      F(ENC_NODE + 4), F(ENC_NODE + 5), F(ENC_NODE + 6), F(ENC_NODE + 7),
      F(ENC_NODE + 9) /*g*/, F(ENC_NODE + 8) /*be*/, nf, N);

  // ---- 10 message-passing steps ----
  const int* src = (const int*)d_in[IN_EIDX];  // edge_index[0]
  for (int s = 0; s < 10; ++s) {
    hipMemsetAsync(agg, 0, (size_t)N * 128 * sizeof(float), stream);
    k_proc_edge<<<blocksE, 128, SMEM_PROC, stream>>>(
        ef, nf, agg, src, WP(o_pE[s]),
        F(PROC_EDGE + 4) + (size_t)s * 128, F(PROC_EDGE + 5) + (size_t)s * 128,
        F(PROC_EDGE + 6) + (size_t)s * 128, F(PROC_EDGE + 7) + (size_t)s * 128,
        F(PROC_EDGE + 9) + (size_t)s * 128 /*g*/,
        F(PROC_EDGE + 8) + (size_t)s * 128 /*be*/, E);
    k_proc_node<<<blocksN, 128, SMEM_PROC, stream>>>(
        nf, agg, WP(o_pN[s]),
        F(PROC_NODE + 4) + (size_t)s * 128, F(PROC_NODE + 5) + (size_t)s * 128,
        F(PROC_NODE + 6) + (size_t)s * 128, F(PROC_NODE + 7) + (size_t)s * 128,
        F(PROC_NODE + 9) + (size_t)s * 128 /*g*/,
        F(PROC_NODE + 8) + (size_t)s * 128 /*be*/, N);
  }

  // ---- decoder ----
  k_decode<<<blocksN, 128, SMEM_ENC, stream>>>(nf, WP(o_dec), F(DEC + 4),
                                               F(DEC + 5), F(DEC + 6),
                                               F(DEC + 7), (float*)d_out, N);
}
